// CrossScanAttention_45071386805014
// MI455X (gfx1250) — compile-verified, matrix operand order checked by reading, not run on hardware
//
#include <hip/hip_runtime.h>
#include <hip/hip_bf16.h>
#include <stdint.h>

// ---------------------------------------------------------------- constants
#define BB      16
#define CC      768
#define LL      768           // sequence length == C
#define NTOK    (BB*LL)       // 12288 tokens
#define DMODEL  64
#define DINNER  128
#define DSTATE  16
#define DTRANK  4
#define DCONV   4
#define DBCPAD  64            // xproj output padded 36 -> 64
#define LN_EPS  1e-5f

typedef __attribute__((ext_vector_type(16))) _Float16 v16h;
typedef __attribute__((ext_vector_type(8)))  float    v8f;

#ifndef __has_builtin
#define __has_builtin(x) 0
#endif
#if __has_builtin(__builtin_amdgcn_global_load_async_to_lds_b128) && \
    __has_builtin(__builtin_amdgcn_s_wait_asynccnt)
#define HAVE_ASYNC_LDS 1
#else
#define HAVE_ASYNC_LDS 0
#endif

// ---------------------------------------------------------------- helpers
__device__ __forceinline__ float dev_silu(float x)    { return x / (1.f + __expf(-x)); }
__device__ __forceinline__ float dev_softplus(float x){ return (x > 20.f) ? x : __logf(1.f + __expf(x)); }

// ------------------------------------------------------- weight conversion
__global__ void cvt_f16_kernel(const float* __restrict__ src, _Float16* __restrict__ dst, int n) {
    int i = blockIdx.x * blockDim.x + threadIdx.x;
    if (i < n) dst[i] = (_Float16)src[i];
}

// xproj_w (36 x 128) -> fp16 padded (64 x 128), zero rows >= 36
__global__ void pad_xproj_kernel(const float* __restrict__ w, _Float16* __restrict__ dst) {
    int i = blockIdx.x * blockDim.x + threadIdx.x;      // 64*128
    if (i >= DBCPAD * DINNER) return;
    int row = i >> 7, col = i & 127;
    dst[i] = (row < DTRANK + 2 * DSTATE) ? (_Float16)w[row * DINNER + col] : (_Float16)0.f;
}

// ----------------------------------------- pool(4x4 mean) + pre proj + LN
__global__ __launch_bounds__(64)
void pool_pre_ln_kernel(const float* __restrict__ img1, const float* __restrict__ img2,
                        const float* __restrict__ pre_w, const float* __restrict__ pre_b,
                        const float* __restrict__ ln_g,  const float* __restrict__ ln_b,
                        _Float16* __restrict__ x1h, _Float16* __restrict__ x2h) {
    __shared__ float pooled[16];
    __shared__ float yv[64];
    const int idx   = blockIdx.x;           // 0 .. 2*B*C-1
    const int which = idx / (BB * CC);
    const int r     = idx % (BB * CC);      // b*C + c
    const float* src = (which ? img2 : img1) + (size_t)r * 1024;
    const int t = threadIdx.x;
    if (t < 16) {                           // one 8x8 block mean per thread
        int bh = t >> 2, bw = t & 3;
        float s = 0.f;
        #pragma unroll
        for (int i = 0; i < 8; ++i)
            #pragma unroll
            for (int j = 0; j < 8; ++j)
                s += src[(bh * 8 + i) * 32 + bw * 8 + j];
        pooled[t] = s * (1.f / 64.f);
    }
    __syncthreads();
    float y = pre_b[t];
    #pragma unroll
    for (int k = 0; k < 16; ++k) y += pooled[k] * pre_w[t * 16 + k];
    yv[t] = y;
    __syncthreads();
    float mu = 0.f;
    #pragma unroll
    for (int k = 0; k < 64; ++k) mu += yv[k];
    mu *= (1.f / 64.f);
    float var = 0.f;
    #pragma unroll
    for (int k = 0; k < 64; ++k) { float d = yv[k] - mu; var += d * d; }
    var *= (1.f / 64.f);
    float o = (y - mu) * rsqrtf(var + LN_EPS) * ln_g[t] + ln_b[t];
    (which ? x2h : x1h)[(size_t)r * 64 + t] = (_Float16)o;
}

// ------------------------------------------------------------- WMMA GEMM
// Cout[M x Npad] = A[M x KT](f16) * W[Npad x KT]^T(f16), f32 accumulate.
// Block: 256 threads = 8 waves as 4(M) x 2(N); block tile 64M x 32N.
// Both tiles are fully contiguous global regions -> flat async b128 copy to LDS.
// Fragment gather implements the CDNA5 16-bit A/B VGPR layout (ISA 7.12.2):
//  lane<16 holds K {0..7,16..23}, lane>=16 holds K {8..15,24..31} per 32-K step.
template <int KT>
__global__ __launch_bounds__(256)
void gemm_wmma_kernel(const uint32_t* __restrict__ A_u,  // M x (KT/2) u32
                      const uint32_t* __restrict__ W_u,  // Npad x (KT/2) u32
                      float* __restrict__ Cout, int Npad) {
    constexpr int KW = KT / 2;
    __shared__ uint32_t As[64 * KW];
    __shared__ uint32_t Ws[32 * KW];
    const int tid = threadIdx.x;
    const int m0  = blockIdx.x * 64;
    const int n0  = blockIdx.y * 32;
    const uint32_t* Ag = A_u + (size_t)m0 * KW;   // 64*KW u32, contiguous
    const uint32_t* Wg = W_u + (size_t)n0 * KW;   // 32*KW u32, contiguous

    __builtin_prefetch(Ag + 64 * KW, 0, 1);       // global_prefetch_b8 (next tile)

#if HAVE_ASYNC_LDS
    {   // async DMA: global -> LDS in 16B chunks, bypassing VGPRs
        typedef int v4i_async __attribute__((vector_size(16)));
        typedef __attribute__((address_space(1))) v4i_async* gp4;  // global int4*
        typedef __attribute__((address_space(3))) v4i_async* lp4;  // LDS int4*
        #pragma unroll
        for (int i = tid; i < 64 * KW / 4; i += 256)
            __builtin_amdgcn_global_load_async_to_lds_b128(
                (gp4)(Ag + i * 4), (lp4)(&As[i * 4]), 0, 0);
        #pragma unroll
        for (int i = tid; i < 32 * KW / 4; i += 256)
            __builtin_amdgcn_global_load_async_to_lds_b128(
                (gp4)(Wg + i * 4), (lp4)(&Ws[i * 4]), 0, 0);
        __builtin_amdgcn_s_wait_asynccnt(0);
    }
#else
    for (int i = tid; i < 64 * KW; i += 256) As[i] = Ag[i];
    for (int i = tid; i < 32 * KW; i += 256) Ws[i] = Wg[i];
#endif
    __syncthreads();

    const int w      = tid >> 5;
    const int lane   = tid & 31;
    const int wm     = (w & 3) * 16;      // M subtile within block
    const int wn     = (w >> 2) * 16;     // N subtile within block
    const int hi     = (lane >> 4) & 1;   // lanes 16..31
    const int row    = lane & 15;

    v8f c = {};
    #pragma unroll
    for (int kk = 0; kk < KT; kk += 32) {
        union { uint32_t u[8]; v16h v; } af, bf;
        #pragma unroll
        for (int vv = 0; vv < 8; ++vv) {
            int k = kk + (vv < 4 ? vv * 2 : 16 + (vv - 4) * 2) + hi * 8;
            af.u[vv] = As[(wm + row) * KW + (k >> 1)];
            bf.u[vv] = Ws[(wn + row) * KW + (k >> 1)];
        }
        c = __builtin_amdgcn_wmma_f32_16x16x32_f16(false, af.v, false, bf.v,
                                                   (short)0, c, false, false);
    }
    // C/D layout: VGPR r -> M = r + 8*hi, N = lane%16
    const int col = n0 + wn + row;
    #pragma unroll
    for (int r = 0; r < 8; ++r)
        Cout[(size_t)(m0 + wm + r + hi * 8) * Npad + col] = c[r];
}

// ------------------------------- causal dwconv + SiLU, and silu(z) gate
__global__ void conv_silu_kernel(const float* __restrict__ xz,   // (B,L,256)
                                 const float* __restrict__ cw,   // (128,4)
                                 const float* __restrict__ cb,   // (128)
                                 float* __restrict__ xl, _Float16* __restrict__ xlh,
                                 float* __restrict__ zsil) {
    int idx = blockIdx.x * blockDim.x + threadIdx.x;             // b*L*128
    if (idx >= NTOK * DINNER) return;
    int d = idx & 127;
    int l = (idx >> 7) % LL;
    int b = idx / (LL * DINNER);
    const float* base = xz + (size_t)b * LL * 256 + d;
    float acc = cb[d];
    #pragma unroll
    for (int k = 0; k < DCONV; ++k) {
        int ll = l - (DCONV - 1) + k;
        if (ll >= 0) acc += cw[d * DCONV + k] * base[(size_t)ll * 256];
    }
    float s = dev_silu(acc);
    xl[idx]  = s;
    xlh[idx] = (_Float16)s;
    zsil[idx] = dev_silu(base[(size_t)l * 256 + DINNER]);        // z channel d+128
}

// -------------------------------------- dt = softplus(dt_raw @ dtw^T + b)
__global__ void dt_kernel(const float* __restrict__ dbc,  // (NTOK,64): dt_raw in [0:4]
                          const float* __restrict__ dtw,  // (128,4)
                          const float* __restrict__ dtb,  // (128)
                          float* __restrict__ dt) {       // (NTOK,128)
    int idx = blockIdx.x * blockDim.x + threadIdx.x;
    if (idx >= NTOK * DINNER) return;
    int d   = idx & 127;
    int tok = idx >> 7;
    const float* db = dbc + (size_t)tok * DBCPAD;
    float v = dtb[d] + dtw[d * 4] * db[0] + dtw[d * 4 + 1] * db[1]
                     + dtw[d * 4 + 2] * db[2] + dtw[d * 4 + 3] * db[3];
    dt[idx] = dev_softplus(v);
}

// ----------------------------------------------------------- reverse copy
__global__ void reverse_kernel(const float* __restrict__ src, float* __restrict__ dst) {
    int i = blockIdx.x * blockDim.x + threadIdx.x;               // B*L*256
    if (i >= NTOK * 256) return;
    int cch = i & 255;
    int l   = (i >> 8) % LL;
    int b   = i / (LL * 256);
    dst[i] = src[((size_t)b * LL + (LL - 1 - l)) * 256 + cch];
}

// -------------------------------------------------------- selective scan
// block: 256 threads = 16 d x 16 s ; grid: B * (128/16) = 128 blocks.
// Serial loop is minimal: one exp + FMA + 4-step shuffle tree per step.
__global__ __launch_bounds__(256)
void scan_kernel(const float* __restrict__ xl,     // (B,L,128) conv+silu
                 const float* __restrict__ dbc,    // (B,L,64): B[4:20] C[20:36]
                 const float* __restrict__ dt,     // (B,L,128) softplus'ed
                 const float* __restrict__ zsil,   // (B,L,128) silu(z)
                 const float* __restrict__ A_log, const float* __restrict__ Dp,
                 float* __restrict__ ysum, int rev, int accum) {
    const int t  = threadIdx.x;
    const int s  = t & 15, dl = t >> 4;
    const int b  = blockIdx.x >> 3;
    const int d  = (blockIdx.x & 7) * 16 + dl;
    const float A  = -__expf(A_log[d * DSTATE + s]);
    const float Dd = Dp[d];
    const size_t baseTok = (size_t)b * LL;
    float h = 0.f;
    for (int l = 0; l < LL; ++l) {
        const float* db = dbc + (baseTok + l) * DBCPAD;
        float dtl = dt[(baseTok + l) * DINNER + d];
        float Bm = db[4 + s], Cm = db[20 + s];
        float xv = xl[(baseTok + l) * DINNER + d];
        h = __expf(dtl * A) * h + (dtl * Bm) * xv;
        float p = h * Cm;
        p += __shfl_xor(p, 1, 32);                 // reduce over the 16 s lanes
        p += __shfl_xor(p, 2, 32);
        p += __shfl_xor(p, 4, 32);
        p += __shfl_xor(p, 8, 32);
        if (s == 0) {
            float y = (p + xv * Dd) * zsil[(baseTok + l) * DINNER + d];
            int lo = rev ? (LL - 1 - l) : l;
            float* dst = ysum + (baseTok + lo) * DINNER + d;
            *dst = accum ? (*dst + y) : y;
        }
    }
}

// ----------------------------------------------------------- post head
__global__ void post_kernel(const float* __restrict__ outm,      // (NTOK,64)
                            const float* __restrict__ pw, const float* __restrict__ pb,
                            float* __restrict__ out) {
    int i = blockIdx.x * blockDim.x + threadIdx.x;
    if (i >= NTOK) return;
    const float* row = outm + (size_t)i * DMODEL;
    float acc = pb[0];
    #pragma unroll
    for (int k = 0; k < DMODEL; ++k) acc += row[k] * pw[k];
    out[i] = 1.f / (1.f + __expf(-0.5f * acc)) + 1e-6f;
}

// ---------------------------------------------------------------- launch
extern "C" void kernel_launch(void* const* d_in, const int* in_sizes, int n_in,
                              void* d_out, int out_size, void* d_ws, size_t ws_size,
                              hipStream_t stream) {
    (void)in_sizes; (void)n_in; (void)out_size; (void)ws_size;
    const float* img1       = (const float*)d_in[0];
    const float* img2       = (const float*)d_in[1];
    const float* pre_w      = (const float*)d_in[2];
    const float* pre_b      = (const float*)d_in[3];
    const float* ln_g       = (const float*)d_in[4];
    const float* ln_b       = (const float*)d_in[5];
    const float* in_proj_w  = (const float*)d_in[6];
    const float* in_proj_sw = (const float*)d_in[7];
    const float* out_proj_w = (const float*)d_in[8];
    const float* post_w     = (const float*)d_in[9];
    const float* post_b     = (const float*)d_in[10];
    const float *conv_w[3], *conv_b[3], *xproj_w[3], *dtw[3], *dtb[3], *Alog[3], *Dvec[3];
    for (int t = 0; t < 3; ++t) {
        int base = 11 + t * 7;
        conv_w[t]  = (const float*)d_in[base + 0];
        conv_b[t]  = (const float*)d_in[base + 1];
        xproj_w[t] = (const float*)d_in[base + 2];
        dtw[t]     = (const float*)d_in[base + 3];
        dtb[t]     = (const float*)d_in[base + 4];
        Alog[t]    = (const float*)d_in[base + 5];
        Dvec[t]    = (const float*)d_in[base + 6];
    }

    // bump allocator on workspace
    size_t off = 0;
    auto alloc = [&](size_t bytes) -> void* {
        void* p = (char*)d_ws + off;
        off += (bytes + 255) & ~(size_t)255;
        return p;
    };
    _Float16* x1h   = (_Float16*)alloc((size_t)NTOK * 64 * 2);
    _Float16* x2h   = (_Float16*)alloc((size_t)NTOK * 64 * 2);
    float*    xz    = (float*)alloc((size_t)NTOK * 256 * 4);
    float*    xzs   = (float*)alloc((size_t)NTOK * 256 * 4);
    float*    xzr   = (float*)alloc((size_t)NTOK * 256 * 4);
    _Float16* wh_in   = (_Float16*)alloc((size_t)256 * 64 * 2);
    _Float16* wh_ins  = (_Float16*)alloc((size_t)256 * 64 * 2);
    _Float16* wh_out  = (_Float16*)alloc((size_t)64 * 128 * 2);
    _Float16* wh_xp[3];
    for (int t = 0; t < 3; ++t) wh_xp[t] = (_Float16*)alloc((size_t)DBCPAD * DINNER * 2);
    float*    xl    = (float*)alloc((size_t)NTOK * DINNER * 4);
    _Float16* xlh   = (_Float16*)alloc((size_t)NTOK * DINNER * 2);
    float*    zsil  = (float*)alloc((size_t)NTOK * DINNER * 4);
    float*    dbc   = (float*)alloc((size_t)NTOK * DBCPAD * 4);
    float*    dtb_f = (float*)alloc((size_t)NTOK * DINNER * 4);
    float*    ysum  = (float*)alloc((size_t)NTOK * DINNER * 4);
    _Float16* ysumh = (_Float16*)alloc((size_t)NTOK * DINNER * 2);
    float*    outm  = (float*)alloc((size_t)NTOK * DMODEL * 4);

    // 1) weights -> fp16 (+ padded xproj)
    cvt_f16_kernel<<<(256 * 64 + 255) / 256, 256, 0, stream>>>(in_proj_w,  wh_in,  256 * 64);
    cvt_f16_kernel<<<(256 * 64 + 255) / 256, 256, 0, stream>>>(in_proj_sw, wh_ins, 256 * 64);
    cvt_f16_kernel<<<(64 * 128 + 255) / 256, 256, 0, stream>>>(out_proj_w, wh_out, 64 * 128);
    for (int t = 0; t < 3; ++t)
        pad_xproj_kernel<<<(DBCPAD * DINNER + 255) / 256, 256, 0, stream>>>(xproj_w[t], wh_xp[t]);

    // 2) pool + pre + LN for both images
    pool_pre_ln_kernel<<<2 * BB * CC, 64, 0, stream>>>(img1, img2, pre_w, pre_b, ln_g, ln_b,
                                                       x1h, x2h);

    // 3) in_proj GEMMs: (12288 x 64) * (256 x 64)^T -> (12288 x 256)
    gemm_wmma_kernel<64><<<dim3(NTOK / 64, 256 / 32), 256, 0, stream>>>(
        (const uint32_t*)x1h, (const uint32_t*)wh_in, xz, 256);
    gemm_wmma_kernel<64><<<dim3(NTOK / 64, 256 / 32), 256, 0, stream>>>(
        (const uint32_t*)x2h, (const uint32_t*)wh_ins, xzs, 256);

    // 4) reversed copy for backward branch
    reverse_kernel<<<(NTOK * 256 + 255) / 256, 256, 0, stream>>>(xz, xzr);

    // 5) three SSM branches: f (xz), b (xzr, reversed output), s (xzs)
    const float* srcs[3] = { xz, xzr, xzs };
    for (int t = 0; t < 3; ++t) {
        conv_silu_kernel<<<(NTOK * DINNER + 255) / 256, 256, 0, stream>>>(
            srcs[t], conv_w[t], conv_b[t], xl, xlh, zsil);
        // xproj: (12288 x 128) * (64 x 128)^T -> (12288 x 64) [cols 0..35 live]
        gemm_wmma_kernel<128><<<dim3(NTOK / 64, DBCPAD / 32), 256, 0, stream>>>(
            (const uint32_t*)xlh, (const uint32_t*)wh_xp[t], dbc, DBCPAD);
        dt_kernel<<<(NTOK * DINNER + 255) / 256, 256, 0, stream>>>(dbc, dtw[t], dtb[t], dtb_f);
        scan_kernel<<<BB * (DINNER / 16), 256, 0, stream>>>(
            xl, dbc, dtb_f, zsil, Alog[t], Dvec[t], ysum,
            /*rev=*/(t == 1) ? 1 : 0, /*accum=*/(t == 0) ? 0 : 1);
    }

    // 6) out_proj: (12288 x 128) * (64 x 128)^T -> (12288 x 64)
    cvt_f16_kernel<<<(NTOK * DINNER + 255) / 256, 256, 0, stream>>>(ysum, ysumh, NTOK * DINNER);
    gemm_wmma_kernel<128><<<dim3(NTOK / 64, 64 / 32), 256, 0, stream>>>(
        (const uint32_t*)ysumh, (const uint32_t*)wh_out, outm, 64);

    // 7) post head -> d_out (B*C floats)
    post_kernel<<<(NTOK + 255) / 256, 256, 0, stream>>>(outm, post_w, post_b, (float*)d_out);
}
